// NeRF_22617297780763
// MI455X (gfx1250) — matrix operand order repaired and checked by reading, not compiled, not run on hardware
//
#include <hip/hip_runtime.h>
#include <hip/hip_bf16.h>

// ---------------------------------------------------------------------------
// NeRF MLP inference for MI455X (gfx1250, wave32, WMMA).
//   - repack_kernel: f32 weights -> f16 WMMA-B-fragment layout in d_ws
//   - nerf_kernel:   128 points / block, f16 activations in LDS,
//                    v_wmma_f32_16x16x32_f16 for all wide layers.
// ---------------------------------------------------------------------------

typedef __attribute__((ext_vector_type(16))) _Float16 v16h;
typedef __attribute__((ext_vector_type(8)))  _Float16 v8h;
typedef __attribute__((ext_vector_type(8)))  float    v8f;

#define MB    128        // points per block
#define ASTR  328        // LDS activation row stride in halves (164 dwords, conflict-free)
#define NPTS  (4096 * 128)

union AFrag { v16h v; v8h h[2]; };

struct Params {
  const float* pos;
  const float* dir;
  const float* b[8];          // trunk biases [256]
  const float* Wd;            // [256]
  const float* bd;            // [1]
  const float* bc0;           // [128]
  const float* Wc1;           // [128*3]
  const float* bc1;           // [3]
  const _Float16* wf[9];      // repacked weights: trunk 0..7, color0 = 8
  float* out_density;         // [NPTS]
  float* out_color;           // [NPTS*3]
};

// --------------------------- weight repack ---------------------------------
// dst fragment (nt, kt) holds 32 lanes x 16 halves contiguous (512 halves):
//   lane l, half h  ->  W[kt*32 + (l<16 ? 0 : 16) + h][nt*16 + (l & 15)]
// K padded with zeros up to Kpad (multiple of 32).
__global__ void repack_kernel(const float* __restrict__ src, _Float16* __restrict__ dst,
                              int Kreal, int Kpad, int N) {
  const int t = blockIdx.x * blockDim.x + threadIdx.x;
  const int total = Kpad * N;
  if (t >= total) return;
  const int ksteps = Kpad >> 5;
  const int frag   = t >> 9;
  const int within = t & 511;
  const int lane   = within >> 4;
  const int h      = within & 15;
  const int nt     = frag / ksteps;
  const int kt     = frag - nt * ksteps;
  const int n      = nt * 16 + (lane & 15);
  const int k      = kt * 32 + ((lane < 16) ? 0 : 16) + h;
  const float v    = (k < Kreal) ? src[(size_t)k * N + n] : 0.0f;
  dst[t] = (_Float16)v;
}

// ----------------------- WMMA GEMM + bias + relu ---------------------------
// src/dst: LDS activation buffers [MB][ASTR] (f16).  Each wave computes NT
// N-tiles of 16 for all 8 M-tiles.  K = ksteps*32.
template<int NT>
__device__ __forceinline__ void gemm_relu(const _Float16* src, _Float16* dst,
                                          const _Float16* __restrict__ wf,
                                          const float* __restrict__ bias,
                                          int ksteps) {
  const int tid  = threadIdx.x;
  const int lane = tid & 31;
  const int wave = tid >> 5;
  const int l15  = lane & 15;
  const int koff = (lane < 16) ? 0 : 16;  // A-matrix K sub-offset (contiguous, matches B repack)
  const int roff = (lane < 16) ? 0 : 8;   // C/D row offset per ISA layout

  v8f acc[8][NT];
  #pragma unroll
  for (int m = 0; m < 8; ++m)
    #pragma unroll
    for (int j = 0; j < NT; ++j)
      #pragma unroll
      for (int r = 0; r < 8; ++r) acc[m][j][r] = 0.0f;

  for (int kt = 0; kt < ksteps; ++kt) {
    v16h bf[NT];
    #pragma unroll
    for (int j = 0; j < NT; ++j) {
      const int nt = wave * NT + j;
      bf[j] = *(const v16h*)(wf + (size_t)(nt * ksteps + kt) * 512 + lane * 16);
    }
    const int kb = kt * 32 + koff;
    #pragma unroll
    for (int m = 0; m < 8; ++m) {
      const _Float16* ap = src + (m * 16 + l15) * ASTR + kb;
      AFrag a;
      a.h[0] = *(const v8h*)(ap);        // K = kb .. kb+7
      a.h[1] = *(const v8h*)(ap + 8);    // K = kb+8 .. kb+15
      #pragma unroll
      for (int j = 0; j < NT; ++j)
        acc[m][j] = __builtin_amdgcn_wmma_f32_16x16x32_f16(
            false, a.v, false, bf[j], (short)0, acc[m][j], false, false);
    }
  }

  #pragma unroll
  for (int j = 0; j < NT; ++j) {
    const int n  = (wave * NT + j) * 16 + l15;
    const float bv = bias[n];
    #pragma unroll
    for (int m = 0; m < 8; ++m) {
      #pragma unroll
      for (int r = 0; r < 8; ++r) {
        float v = acc[m][j][r] + bv;
        v = v > 0.0f ? v : 0.0f;
        dst[(m * 16 + r + roff) * ASTR + n] = (_Float16)v;
      }
    }
  }
}

// ------------------------------ main kernel --------------------------------
__global__ void __launch_bounds__(256) nerf_kernel(Params p) {
  __shared__ __align__(16) _Float16 smem[2 * MB * ASTR];   // ~164 KB
  _Float16* bufA = smem;
  _Float16* bufB = smem + MB * ASTR;

  const int tid = threadIdx.x;
  const int p0  = blockIdx.x * MB;

  // ---- positional encoding: bufA[:,0:60] (layer-0 input), bufB[:,256:316] (skip)
  for (int i = tid; i < MB * 3; i += 256) {
    const int pt = i / 3, ax = i - pt * 3;
    float f = p.pos[(size_t)(p0 + pt) * 3 + ax];
    #pragma unroll
    for (int r = 0; r < 10; ++r) {
      const float s = __sinf(f), c = __cosf(f);
      const int cs = ax * 20 + r;
      bufA[pt * ASTR + cs]            = (_Float16)s;
      bufA[pt * ASTR + cs + 10]       = (_Float16)c;
      bufB[pt * ASTR + 256 + cs]      = (_Float16)s;
      bufB[pt * ASTR + 256 + cs + 10] = (_Float16)c;
      f += f;   // next octave
    }
  }
  // zero-pad cols 60..63 (K pad of layer 0 and of skip concat)
  for (int i = tid; i < MB * 4; i += 256) {
    const int pt = i >> 2, c = 60 + (i & 3);
    bufA[pt * ASTR + c]       = (_Float16)0.0f;
    bufB[pt * ASTR + 256 + c] = (_Float16)0.0f;
  }
  __syncthreads();

  // ---- trunk (ping-pong A<->B); skip concat is free: bufB[:,256:320] = embed_pos
  __builtin_prefetch(p.wf[1], 0, 1);
  gemm_relu<2>(bufA, bufB, p.wf[0], p.b[0], 2);   __syncthreads();
  __builtin_prefetch(p.wf[2], 0, 1);
  gemm_relu<2>(bufB, bufA, p.wf[1], p.b[1], 8);   __syncthreads();
  __builtin_prefetch(p.wf[3], 0, 1);
  gemm_relu<2>(bufA, bufB, p.wf[2], p.b[2], 8);   __syncthreads();
  __builtin_prefetch(p.wf[4], 0, 1);
  gemm_relu<2>(bufB, bufA, p.wf[3], p.b[3], 8);   __syncthreads();
  __builtin_prefetch(p.wf[5], 0, 1);
  gemm_relu<2>(bufA, bufB, p.wf[4], p.b[4], 8);   __syncthreads();
  __builtin_prefetch(p.wf[6], 0, 1);
  gemm_relu<2>(bufB, bufA, p.wf[5], p.b[5], 10);  __syncthreads();   // K=320 (skip)
  __builtin_prefetch(p.wf[7], 0, 1);
  gemm_relu<2>(bufA, bufB, p.wf[6], p.b[6], 8);   __syncthreads();
  __builtin_prefetch(p.wf[8], 0, 1);
  gemm_relu<2>(bufB, bufA, p.wf[7], p.b[7], 8);   __syncthreads();
  // hx in bufA[:, 0:256]

  // ---- density head (256 -> 1) with plain dots; 128 threads, 1 point each
  if (tid < MB) {
    float s = p.bd[0];
    for (int k = 0; k < 256; ++k)
      s += (float)bufA[tid * ASTR + k] * p.Wd[k];
    p.out_density[p0 + tid] = s;
  }

  // ---- direction encoding into bufA[:,256:280], zeros 280..287 (color K pad)
  for (int i = tid; i < MB * 3; i += 256) {
    const int pt = i / 3, ax = i - pt * 3;
    float f = p.dir[(size_t)(p0 + pt) * 3 + ax];
    #pragma unroll
    for (int r = 0; r < 4; ++r) {
      const int cs = 256 + ax * 8 + r;
      bufA[pt * ASTR + cs]     = (_Float16)__sinf(f);
      bufA[pt * ASTR + cs + 4] = (_Float16)__cosf(f);
      f += f;
    }
  }
  for (int i = tid; i < MB * 8; i += 256) {
    const int pt = i >> 3, c = 280 + (i & 7);
    bufA[pt * ASTR + c] = (_Float16)0.0f;
  }
  __syncthreads();

  // ---- color hidden: K=288 (9 ksteps), N=128 -> each wave 1 N-tile
  gemm_relu<1>(bufA, bufB, p.wf[8], p.bc0, 9);
  __syncthreads();

  // ---- color head (128 -> 3)
  for (int i = tid; i < MB * 3; i += 256) {
    const int pt = i / 3, c = i - pt * 3;
    float s = p.bc1[c];
    for (int k = 0; k < 128; ++k)
      s += (float)bufB[pt * ASTR + k] * p.Wc1[k * 3 + c];
    p.out_color[(size_t)(p0 + pt) * 3 + c] = s;
  }
}

// ------------------------------- launcher ----------------------------------
extern "C" void kernel_launch(void* const* d_in, const int* in_sizes, int n_in,
                              void* d_out, int out_size, void* d_ws, size_t ws_size,
                              hipStream_t stream) {
  (void)in_sizes; (void)n_in; (void)out_size; (void)ws_size;
  _Float16* ws = (_Float16*)d_ws;

  struct LayerDesc { int in_idx, Kreal, Kpad, N; };
  const LayerDesc L[9] = {
    { 2,  60,  64, 256},   // W0
    { 4, 256, 256, 256},   // W1
    { 6, 256, 256, 256},   // W2
    { 8, 256, 256, 256},   // W3
    {10, 256, 256, 256},   // W4
    {12, 316, 320, 256},   // W5 (skip input 256+60)
    {14, 256, 256, 256},   // W6
    {16, 256, 256, 256},   // W7
    {20, 280, 288, 128},   // Wc0 (256+24)
  };

  Params prm;
  size_t off = 0;
  for (int i = 0; i < 9; ++i) {
    const int total = L[i].Kpad * L[i].N;
    repack_kernel<<<(total + 255) / 256, 256, 0, stream>>>(
        (const float*)d_in[L[i].in_idx], ws + off, L[i].Kreal, L[i].Kpad, L[i].N);
    prm.wf[i] = ws + off;
    off += (size_t)total;
  }

  prm.pos = (const float*)d_in[0];
  prm.dir = (const float*)d_in[1];
  for (int i = 0; i < 8; ++i) prm.b[i] = (const float*)d_in[3 + 2 * i];
  prm.Wd  = (const float*)d_in[18];
  prm.bd  = (const float*)d_in[19];
  prm.bc0 = (const float*)d_in[21];
  prm.Wc1 = (const float*)d_in[22];
  prm.bc1 = (const float*)d_in[23];
  prm.out_density = (float*)d_out;
  prm.out_color   = (float*)d_out + (size_t)NPTS;

  nerf_kernel<<<NPTS / MB, 256, 0, stream>>>(prm);
}
